// BP_HFNN_70042326663683
// MI455X (gfx1250) — matrix operand order repaired and checked by reading, not compiled, not run on hardware
//
#include <hip/hip_runtime.h>
#include <hip/hip_bf16.h>

typedef __attribute__((ext_vector_type(16))) _Float16 v16h;
typedef __attribute__((ext_vector_type(8)))  float    v8f;
typedef __attribute__((ext_vector_type(4)))  float    v4f;

// Problem sizes (fixed by reference)
#define N_BR    8
#define N_RULES 32
#define N_FEA   16
#define N_BATCH 32768

// Workspace layout (bytes):
//  [0, 32768)        : packed f16 B-operands: 32 matrices (matId = k*4 + which*2 + t),
//                      each 32 lanes x 16 halfwords, stored in WMMA B per-lane order.
//  [32768, 33792)    : float Cc[8][32]   = sum_f mu^2/(2 sigma^2)   (expo C accumulator init)
//  [33792, 34816)    : float Bias[8][32] = w3[k][r][16]             (conq C accumulator init)
#define WS_B_BYTES   32768
#define WS_CC_OFF    32768
#define WS_BIAS_OFF  33792

// ---------------------------------------------------------------------------
// Prep: transform params into WMMA-B-layout f16 operands + per-rule constants.
// B element layout (16-bit B, 32x16, 8 VGPRs): lane L holds column n = L%16;
// element e (halfword) of the lane's 16 halfwords maps to k = e + 8*(L/16) for e<8, k = 16 + 8*(L/16) + (e-8) for e>=8.
// which==0 (expo):  k<16 -> a = 1/(2 s^2);  k>=16 -> -mu/s^2   (feature f = k-16 resp. k)
// which==1 (conq):  k<16 -> 0;              k>=16 -> w3[r][k-16]
// ---------------------------------------------------------------------------
__global__ __launch_bounds__(256) void hfnn_prep(
    const float* __restrict__ mu, const float* __restrict__ sigma,
    const float* __restrict__ w3, _Float16* __restrict__ wsB,
    float* __restrict__ Cc, float* __restrict__ Bias)
{
    int tid = blockIdx.x * 256 + threadIdx.x;
    if (tid < 1024) {
        const int lane  = tid & 31;
        const int matId = tid >> 5;         // 0..31 = k*4 + which*2 + t
        const int t     = matId & 1;
        const int which = (matId >> 1) & 1;
        const int k     = matId >> 2;
        const int n     = lane & 15;
        const int half  = lane >> 4;
        const int r     = t * 16 + n;
        _Float16* dst = wsB + matId * 512 + lane * 16;
        const float* muR = mu    + (k * N_RULES + r) * N_FEA;
        const float* sgR = sigma + (k * N_RULES + r) * N_FEA;
        const float* w3R = w3    + (k * N_RULES + r) * (N_FEA + 1);
        #pragma unroll
        for (int e = 0; e < 16; ++e) {
            const int kb = (e < 8) ? (e + 8 * half) : (16 + 8 * half + (e - 8));
            float v;
            if (which == 0) {
                if (kb < 16) { float s = sgR[kb]; v = 0.5f / (s * s); }
                else         { int f = kb - 16; float s = sgR[f]; v = -muR[f] / (s * s); }
            } else {
                v = (kb < 16) ? 0.0f : w3R[kb - 16];
            }
            dst[e] = (_Float16)v;
        }
    } else if (tid < 1280) {
        const int idx = tid - 1024;         // k*32 + r
        const int k = idx >> 5, r = idx & 31;
        const float* muR = mu    + (k * N_RULES + r) * N_FEA;
        const float* sgR = sigma + (k * N_RULES + r) * N_FEA;
        float c = 0.f;
        #pragma unroll
        for (int f = 0; f < N_FEA; ++f) {
            float m = muR[f], s = sgR[f];
            c += 0.5f * m * m / (s * s);
        }
        Cc[idx]   = c;
        Bias[idx] = w3[(k * N_RULES + r) * (N_FEA + 1) + N_FEA];
    }
}

// ---------------------------------------------------------------------------
// Main: one wave32 per 16-row batch tile. Per branch k:
//   A = [x^2 | x]  (16x32, f16, per-ISA A layout: lane m = L%16, K-half by L/16)
//   expo tiles: c{0,1} = A x Bexpo{0,1} + Cc     (v_wmma_f32_16x16x32_f16)
//   conq tiles: q{0,1} = A x Bconq{0,1} + bias
// Then rule = exp(-expo), normalize over 32 rules (16-lane shfl_xor reductions),
// tsk = sum(norm*conq), logits += tsk * w5, softmax at the end.
// ---------------------------------------------------------------------------
__global__ __launch_bounds__(256) void hfnn_main(
    const float* __restrict__ data, const _Float16* __restrict__ wsB,
    const float* __restrict__ Cc, const float* __restrict__ Bias,
    const float* __restrict__ w5, const float* __restrict__ b5,
    float* __restrict__ out)
{
    const int wave = (blockIdx.x * blockDim.x + threadIdx.x) >> 5;  // tile id, 0..2047
    const int lane = threadIdx.x & 31;
    const int base = wave * 16;            // first batch row of tile
    const int n    = lane & 15;
    const int half = lane >> 4;
    const int off  = 8 * half;             // K-half offset for A operand

    const v16h* __restrict__ Bv = (const v16h*)wsB;  // 32 v16h per matrix

    float l0[8], l1[8];
    #pragma unroll
    for (int i = 0; i < 8; ++i) { l0[i] = 0.f; l1[i] = 0.f; }

    for (int k = 0; k < N_BR; ++k) {
        // --- load this lane's data row (batch b = base + n), build A operand ---
        const float* row = data + ((size_t)(k * N_BATCH + base + n)) * N_FEA;
        v4f x0 = *(const v4f*)(row + 0);
        v4f x1 = *(const v4f*)(row + 4);
        v4f x2 = *(const v4f*)(row + 8);
        v4f x3 = *(const v4f*)(row + 12);
        float xr[16] = { x0.x, x0.y, x0.z, x0.w, x1.x, x1.y, x1.z, x1.w,
                         x2.x, x2.y, x2.z, x2.w, x3.x, x3.y, x3.z, x3.w };
        v16h a;
        #pragma unroll
        for (int e = 0; e < 8; ++e) {       // K-range [0,16): x^2 terms
            float xv = xr[e + off];
            a[e] = (_Float16)(xv * xv);
        }
        #pragma unroll
        for (int e = 8; e < 16; ++e) {      // K-range [16,32): linear x terms
            a[e] = (_Float16)xr[e - 8 + off];
        }

        // --- pre-packed B operands (L2-resident, 32B vector load per lane) ---
        v16h be0 = Bv[(k * 4 + 0) * 32 + lane];   // expo, rules 0..15
        v16h be1 = Bv[(k * 4 + 1) * 32 + lane];   // expo, rules 16..31
        v16h bq0 = Bv[(k * 4 + 2) * 32 + lane];   // conq, rules 0..15
        v16h bq1 = Bv[(k * 4 + 3) * 32 + lane];   // conq, rules 16..31

        // --- accumulator init: column constants (C/D column = lane%16) ---
        const float cc0 = Cc[k * N_RULES + n];
        const float cc1 = Cc[k * N_RULES + 16 + n];
        const float bb0 = Bias[k * N_RULES + n];
        const float bb1 = Bias[k * N_RULES + 16 + n];
        v8f c0, c1, q0, q1;
        #pragma unroll
        for (int i = 0; i < 8; ++i) { c0[i] = cc0; c1[i] = cc1; q0[i] = bb0; q1[i] = bb1; }

        c0 = __builtin_amdgcn_wmma_f32_16x16x32_f16(false, a, false, be0, (short)0, c0, false, false);
        c1 = __builtin_amdgcn_wmma_f32_16x16x32_f16(false, a, false, be1, (short)0, c1, false, false);
        q0 = __builtin_amdgcn_wmma_f32_16x16x32_f16(false, a, false, bq0, (short)0, q0, false, false);
        q1 = __builtin_amdgcn_wmma_f32_16x16x32_f16(false, a, false, bq1, (short)0, q1, false, false);

        // --- rule activation, normalization, TSK mix ---
        const float w50 = w5[k];            // w5[0][k]
        const float w51 = w5[N_BR + k];     // w5[1][k]
        #pragma unroll
        for (int i = 0; i < 8; ++i) {       // D row m = i + 8*half
            float e0 = __expf(-c0[i]);
            float e1 = __expf(-c1[i]);
            float den = e0 + e1;
            float num = e0 * q0[i] + e1 * q1[i];
            #pragma unroll
            for (int m = 1; m < 16; m <<= 1) {   // reduce over 16 columns (same half)
                den += __shfl_xor(den, m, 32);
                num += __shfl_xor(num, m, 32);
            }
            float tsk = num / den;
            l0[i] += tsk * w50;
            l1[i] += tsk * w51;
        }
    }

    // --- softmax over 2 classes; lanes n==0 of each half write 8 rows each ---
    if (n == 0) {
        const float bb0 = b5[0], bb1 = b5[1];
        #pragma unroll
        for (int i = 0; i < 8; ++i) {
            int b = base + i + 8 * half;
            float a0 = l0[i] + bb0, a1 = l1[i] + bb1;
            float mx = fmaxf(a0, a1);
            float e0 = __expf(a0 - mx), e1 = __expf(a1 - mx);
            float inv = 1.f / (e0 + e1);
            out[2 * b + 0] = e0 * inv;
            out[2 * b + 1] = e1 * inv;
        }
    }
}

// ---------------------------------------------------------------------------
extern "C" void kernel_launch(void* const* d_in, const int* in_sizes, int n_in,
                              void* d_out, int out_size, void* d_ws, size_t ws_size,
                              hipStream_t stream) {
    const float* data  = (const float*)d_in[0];
    const float* mu    = (const float*)d_in[1];
    const float* sigma = (const float*)d_in[2];
    const float* w3    = (const float*)d_in[3];
    const float* w5    = (const float*)d_in[4];
    const float* b5    = (const float*)d_in[5];
    float* out = (float*)d_out;

    _Float16* wsB = (_Float16*)d_ws;
    float* Cc   = (float*)((char*)d_ws + WS_CC_OFF);
    float* Bias = (float*)((char*)d_ws + WS_BIAS_OFF);

    // Pack params into WMMA operand layout (tiny, serialized before main on stream)
    hfnn_prep<<<5, 256, 0, stream>>>(mu, sigma, w3, wsB, Cc, Bias);

    // 2048 batch tiles of 16 rows; 8 waves/block -> 256 blocks
    hfnn_main<<<256, 256, 0, stream>>>(data, wsB, Cc, Bias, w5, b5, out);
}
